// Transformer_60859686584407
// MI455X (gfx1250) — compile-verified, matrix operand order checked by reading, not run on hardware
//
#include <hip/hip_runtime.h>
#include <hip/hip_bf16.h>
#include <math.h>

// ---------------- problem constants (from reference) ----------------
#define WB    4
#define WS_   1024
#define WF    16
#define WD    512
#define WH    8
#define WDH   512
#define WL    6
#define WDFF  2048
#define WNE   256
#define WHALF 256
#define WM    (WB * WS_)   // 4096 rows (B*S)
#define WHD   (WH * WDH)   // 4096
#define ALPHA_RES 1.8612097182041991f   // (2*6)^0.25
#define LN_EPS 1e-5f

typedef __bf16 bf16x16 __attribute__((ext_vector_type(16)));
typedef __bf16 bf16x8  __attribute__((ext_vector_type(8)));
typedef float  f32x8   __attribute__((ext_vector_type(8)));

#define SAS 40   // LDS row stride (32 + 8 pad), 80B keeps 16B alignment

// CDNA5 async global->LDS copy (ASYNCcnt-tracked); fall back to sync copy.
#if __has_builtin(__builtin_amdgcn_global_load_async_to_lds_b128) && \
    __has_builtin(__builtin_amdgcn_s_wait_asynccnt)
#define USE_ASYNC 1
#else
#define USE_ASYNC 0
#endif

// exact parameter types per hipcc diagnostic:
//   param0: int __attribute__((vector_size(16))) __device__(AS1) *
//   param1: AS3 analog
typedef int v4i_gcc __attribute__((vector_size(4 * sizeof(int))));
typedef __attribute__((address_space(1))) v4i_gcc* as1_v4i;
typedef __attribute__((address_space(3))) v4i_gcc* as3_v4i;

__device__ __forceinline__ void cp16g2l(const __bf16* g, __bf16* l)
{
#if USE_ASYNC
    __builtin_amdgcn_global_load_async_to_lds_b128((as1_v4i)g, (as3_v4i)l, 0, 0);
#else
    *(uint4*)l = *(const uint4*)g;
#endif
}

__device__ __forceinline__ void wait_stage()
{
#if USE_ASYNC
    __builtin_amdgcn_s_wait_asynccnt(0);
#endif
    __syncthreads();
}

// =====================================================================
// Batched GEMM:  C = act( (A[M,K] x B^T) * scale + bias )
//   A bf16 row-major [M,K] (lda).  B bf16 stored transposed [N,K] (ldb).
//   Batch z offset = (z/cnt)*outer + (z%cnt)*inner per operand.
//   Outputs: Cf (f32) and/or Cb (bf16); cT=1 -> Cb stored transposed [N,M] (ldc=M).
//   M % 128 == 0, N % 256 == 0, K % 32 == 0.
//   Block: 256 threads = 8 waves (2x4), wave tile 64x64 = 4x4 WMMA.
// =====================================================================
__global__ __launch_bounds__(256)
void gemm_wmma_bf16(const __bf16* __restrict__ A, int lda, long aOut, long aIn, int aCnt,
                    const __bf16* __restrict__ Bm, int ldb, long bOut, long bIn, int bCnt,
                    const float* __restrict__ bias,
                    float* __restrict__ Cf, __bf16* __restrict__ Cb,
                    int ldc, long cOut, long cIn, int cCnt, int cT,
                    int K, float scale, int relu)
{
    const int z = blockIdx.z;
    A  += (long)(z / aCnt) * aOut + (long)(z % aCnt) * aIn;
    Bm += (long)(z / bCnt) * bOut + (long)(z % bCnt) * bIn;
    const long coff = (long)(z / cCnt) * cOut + (long)(z % cCnt) * cIn;

    const int t    = threadIdx.x;
    const int lane = t & 31;
    const int wave = t >> 5;         // 8 waves
    const int wm   = wave >> 2;      // 0..1  (64-row stripes)
    const int wn   = wave & 3;       // 0..3  (64-col stripes)
    const int l16  = lane & 15;
    const int hsel = lane >> 4;      // half-wave select per ISA 16-bit layout
    const int m_blk = blockIdx.y * 128;
    const int n_blk = blockIdx.x * 256;

    // double-buffered LDS tiles: A 128x32, B(transposed) 256x32
    __shared__ __align__(16) __bf16 sA[2 * 128 * SAS];
    __shared__ __align__(16) __bf16 sB[2 * 256 * SAS];

    const int rowF = t & 127;            // A tile row this thread fills
    const int khF  = (t >> 7) << 4;      // 0 or 16 : A k-half

    auto stage = [&](int k0, int b) {
        __bf16* sAb = sA + b * (128 * SAS);
        __bf16* sBb = sB + b * (256 * SAS);
        const __bf16* ga = A + (long)(m_blk + rowF) * lda + k0 + khF;
        cp16g2l(ga,     &sAb[rowF * SAS + khF]);
        cp16g2l(ga + 8, &sAb[rowF * SAS + khF + 8]);
        const __bf16* gb = Bm + (long)(n_blk + t) * ldb + k0;
        __bf16* lb = &sBb[t * SAS];
        cp16g2l(gb,      lb);
        cp16g2l(gb + 8,  lb + 8);
        cp16g2l(gb + 16, lb + 16);
        cp16g2l(gb + 24, lb + 24);
    };

    f32x8 acc[4][4];
    const f32x8 zf = {0.f, 0.f, 0.f, 0.f, 0.f, 0.f, 0.f, 0.f};
#pragma unroll
    for (int i = 0; i < 4; ++i)
#pragma unroll
        for (int j = 0; j < 4; ++j) acc[i][j] = zf;

    stage(0, 0);                 // prologue
    int buf = 0;

    for (int k0 = 0; k0 < K; k0 += 32) {
        wait_stage();            // my async ops done + all waves' tiles visible
        if (k0 + 32 < K) stage(k0 + 32, buf ^ 1);   // overlap next tile with compute

        const __bf16* sAb = sA + buf * (128 * SAS);
        const __bf16* sBb = sB + buf * (256 * SAS);

        // fragments per CDNA5 16-bit layout:
        // lanes 0-15: elems 0..7 = K0..7,  8..15 = K16..23
        // lanes16-31: elems 0..7 = K8..15, 8..15 = K24..31
        bf16x16 af[4];
#pragma unroll
        for (int i = 0; i < 4; ++i) {
            const __bf16* pa = &sAb[(wm * 64 + i * 16 + l16) * SAS + hsel * 8];
            bf16x8 lo = *(const bf16x8*)pa;
            bf16x8 hi = *(const bf16x8*)(pa + 16);
            af[i] = __builtin_shufflevector(lo, hi, 0,1,2,3,4,5,6,7,8,9,10,11,12,13,14,15);
        }
        bf16x16 bfr[4];
#pragma unroll
        for (int j = 0; j < 4; ++j) {
            const __bf16* pb = &sBb[(wn * 64 + j * 16 + l16) * SAS + hsel * 8];
            bf16x8 lo = *(const bf16x8*)pb;
            bf16x8 hi = *(const bf16x8*)(pb + 16);
            bfr[j] = __builtin_shufflevector(lo, hi, 0,1,2,3,4,5,6,7,8,9,10,11,12,13,14,15);
        }
#pragma unroll
        for (int i = 0; i < 4; ++i)
#pragma unroll
            for (int j = 0; j < 4; ++j)
                acc[i][j] = __builtin_amdgcn_wmma_f32_16x16x32_bf16(
                    false, af[i], false, bfr[j], (short)0, acc[i][j], false, false);

        buf ^= 1;
    }

    // ---- epilogue: C layout VGPR e -> row = e + hsel*8, col = l16 ----
#pragma unroll
    for (int i = 0; i < 4; ++i) {
#pragma unroll
        for (int j = 0; j < 4; ++j) {
            const int gc = n_blk + wn * 64 + j * 16 + l16;
            const float bv = bias ? bias[gc] : 0.f;
            if (cT) {
                // transposed bf16 store: 8 consecutive rows -> one 16B store
                const int grb = m_blk + wm * 64 + i * 16 + hsel * 8;
                bf16x8 pk;
#pragma unroll
                for (int e = 0; e < 8; ++e) {
                    float v = acc[i][j][e] * scale + bv;
                    if (relu) v = fmaxf(v, 0.f);
                    pk[e] = (__bf16)v;
                }
                *(bf16x8*)&Cb[coff + (long)gc * ldc + grb] = pk;
            } else {
#pragma unroll
                for (int e = 0; e < 8; ++e) {
                    const int gr = m_blk + wm * 64 + i * 16 + hsel * 8 + e;
                    float v = acc[i][j][e] * scale + bv;
                    if (relu) v = fmaxf(v, 0.f);
                    const long off = coff + (long)gr * ldc + gc;
                    if (Cf) Cf[off] = v;
                    if (Cb) Cb[off] = (__bf16)v;
                }
            }
        }
    }
}

// =====================================================================
// transpose + convert:  src f32 [K,N] (per layer) -> dst bf16 [N,K]
// grid (N/32, K/32, L), block (32,8)
// =====================================================================
__global__ __launch_bounds__(256)
void transpose_f32_to_bf16(const float* __restrict__ src, __bf16* __restrict__ dst,
                           int K, int N)
{
    __shared__ float tile[32][33];
    const int n0 = blockIdx.x * 32;
    const int k0 = blockIdx.y * 32;
    const float* s = src + (long)blockIdx.z * K * N;
    __bf16*      d = dst + (long)blockIdx.z * K * N;
    const int tx = threadIdx.x;      // 32
    const int ty = threadIdx.y;      // 8
#pragma unroll
    for (int i = ty; i < 32; i += 8)
        tile[i][tx] = s[(long)(k0 + i) * N + n0 + tx];
    __syncthreads();
#pragma unroll
    for (int i = ty; i < 32; i += 8)
        d[(long)(n0 + i) * K + k0 + tx] = (__bf16)tile[tx][i];
}

// =====================================================================
// UniEmbedding:  x[b,s,d] = sum_f cos/sin(in*exp(freq)+phase)*w*validM
// =====================================================================
__global__ void embed_kernel(const float* __restrict__ inputs, const int* __restrict__ itype,
                             const float* __restrict__ efreq, const float* __restrict__ ew,
                             const float* __restrict__ ephase,
                             float* __restrict__ x, __bf16* __restrict__ xb)
{
    const long idx = (long)blockIdx.x * blockDim.x + threadIdx.x;
    if (idx >= (long)WM * WHALF) return;
    const int bs = (int)(idx / WHALF);
    const int d  = (int)(idx % WHALF);
    const float* inp = inputs + (long)bs * WF;
    const int*   ty  = itype  + (long)bs * WF;

    float vc = 0.f;
#pragma unroll
    for (int f = 0; f < WF; ++f)
        vc += (ty[f] != 0 && ty[f] != -1) ? 1.f : 0.f;
    const float vm = (vc > 0.f) ? 1.f / (sqrtf(vc) * 1.41421356237f) : 0.f;

    float a0 = 0.f, a1 = 0.f;
#pragma unroll
    for (int f = 0; f < WF; ++f) {
        const int tt = ty[f];
        const bool valid = (tt != 0 && tt != -1);
        const float freq = expf(efreq[(long)tt * WHALF + d]);
        const float ph   = ephase[(long)(f + 1) * WHALF + d];
        const float w    = ew[(long)tt * WHALF + d] * (valid ? vm : 0.f);
        const float dd   = inp[f] * freq + ph;
        a0 += cosf(dd) * w;
        a1 += sinf(dd) * w;
    }
    const long o0 = (long)bs * WD + d;
    const long o1 = o0 + WHALF;
    x[o0] = a0;  x[o1] = a1;
    xb[o0] = (__bf16)a0;  xb[o1] = (__bf16)a1;
}

// =====================================================================
// In-place row softmax over bf16 scores (row length n, one block per row)
// =====================================================================
__global__ __launch_bounds__(128)
void softmax_rows(__bf16* __restrict__ p, int n)
{
    const int t = threadIdx.x;               // 128 threads
    const int per = n / 128;                 // 8
    __bf16* pr = p + (long)blockIdx.x * n;

    float v[8];
    float mx = -3.0e38f;
    for (int i = 0; i < per; ++i) { v[i] = (float)pr[t * per + i]; mx = fmaxf(mx, v[i]); }

    __shared__ float red[128];
    red[t] = mx; __syncthreads();
    for (int s = 64; s > 0; s >>= 1) { if (t < s) red[t] = fmaxf(red[t], red[t + s]); __syncthreads(); }
    mx = red[0]; __syncthreads();

    float sum = 0.f;
    for (int i = 0; i < per; ++i) { v[i] = expf(v[i] - mx); sum += v[i]; }
    red[t] = sum; __syncthreads();
    for (int s = 64; s > 0; s >>= 1) { if (t < s) red[t] += red[t + s]; __syncthreads(); }
    const float inv = 1.f / red[0];

    for (int i = 0; i < per; ++i) pr[t * per + i] = (__bf16)(v[i] * inv);
}

// =====================================================================
// x = LayerNorm(x*alpha + y)*g + b   (row length WD, one block per row)
// =====================================================================
__global__ __launch_bounds__(128)
void add_ln_kernel(float* __restrict__ x, __bf16* __restrict__ xb,
                   const float* __restrict__ y,
                   const float* __restrict__ g, const float* __restrict__ bparm,
                   float alpha)
{
    const int t = threadIdx.x;               // 128 threads, 4 elems each
    const long base = (long)blockIdx.x * WD;
    float tv[4];
    float s = 0.f, s2 = 0.f;
#pragma unroll
    for (int i = 0; i < 4; ++i) {
        const int c = t * 4 + i;
        const float val = x[base + c] * alpha + y[base + c];
        tv[i] = val; s += val; s2 += val * val;
    }
    __shared__ float r1[128], r2[128];
    r1[t] = s; r2[t] = s2; __syncthreads();
    for (int st = 64; st > 0; st >>= 1) {
        if (t < st) { r1[t] += r1[t + st]; r2[t] += r2[t + st]; }
        __syncthreads();
    }
    const float mean = r1[0] * (1.f / WD);
    const float var  = r2[0] * (1.f / WD) - mean * mean;
    const float inv  = rsqrtf(var + LN_EPS);
#pragma unroll
    for (int i = 0; i < 4; ++i) {
        const int c = t * 4 + i;
        const float o = (tv[i] - mean) * inv * g[c] + bparm[c];
        x[base + c] = o;
        xb[base + c] = (__bf16)o;
    }
}

// =====================================================================
// head: mu = x.wout + bout ; sigma = softplus(x.wsig + bsig)
// =====================================================================
__global__ __launch_bounds__(64)
void head_kernel(const float* __restrict__ x,
                 const float* __restrict__ wout, const float* __restrict__ bout,
                 const float* __restrict__ wsig, const float* __restrict__ bsig,
                 float* __restrict__ out)
{
    const int t = threadIdx.x;               // 64 threads, 8 elems each
    const long base = (long)blockIdx.x * WD;
    float m = 0.f, sg = 0.f;
#pragma unroll
    for (int i = 0; i < 8; ++i) {
        const int c = t * 8 + i;
        const float xv = x[base + c];
        m += xv * wout[c];
        sg += xv * wsig[c];
    }
    __shared__ float rm[64], rs[64];
    rm[t] = m; rs[t] = sg; __syncthreads();
    for (int s = 32; s > 0; s >>= 1) {
        if (t < s) { rm[t] += rm[t + s]; rs[t] += rs[t + s]; }
        __syncthreads();
    }
    if (t == 0) {
        const float mu = rm[0] + bout[0];
        const float z  = rs[0] + bsig[0];
        const float sp = (z > 20.f) ? z : log1pf(expf(z));
        out[(long)blockIdx.x * 2 + 0] = mu;
        out[(long)blockIdx.x * 2 + 1] = sp;
    }
}

// =====================================================================
// host side
// =====================================================================
static void launch_gemm(hipStream_t st,
                        const __bf16* A, int lda, long aOut, long aIn, int aCnt,
                        const __bf16* B, int ldb, long bOut, long bIn, int bCnt,
                        const float* bias, float* Cf, __bf16* Cb,
                        int ldc, long cOut, long cIn, int cCnt, int cT,
                        int M, int N, int K, int nbatch, float scale, int relu)
{
    dim3 g(N / 256, M / 128, nbatch), b(256);
    gemm_wmma_bf16<<<g, b, 0, st>>>(A, lda, aOut, aIn, aCnt,
                                    B, ldb, bOut, bIn, bCnt,
                                    bias, Cf, Cb, ldc, cOut, cIn, cCnt, cT,
                                    K, scale, relu);
}

extern "C" void kernel_launch(void* const* d_in, const int* in_sizes, int n_in,
                              void* d_out, int out_size, void* d_ws, size_t ws_size,
                              hipStream_t stream)
{
    (void)in_sizes; (void)n_in; (void)out_size; (void)ws_size;

    const float* inputs = (const float*)d_in[0];
    const int*   itype  = (const int*)  d_in[1];
    const float* efreq  = (const float*)d_in[2];
    const float* ew     = (const float*)d_in[3];
    const float* ephase = (const float*)d_in[4];
    const float* wq = (const float*)d_in[5];  const float* bq = (const float*)d_in[6];
    const float* wk = (const float*)d_in[7];  const float* bk = (const float*)d_in[8];
    const float* wv = (const float*)d_in[9];  const float* bv = (const float*)d_in[10];
    const float* wo = (const float*)d_in[11]; const float* bo = (const float*)d_in[12];
    const float* ln1g = (const float*)d_in[13]; const float* ln1b = (const float*)d_in[14];
    const float* w1 = (const float*)d_in[15]; const float* b1 = (const float*)d_in[16];
    const float* w2 = (const float*)d_in[17]; const float* b2 = (const float*)d_in[18];
    const float* ln2g = (const float*)d_in[19]; const float* ln2b = (const float*)d_in[20];
    const float* wout = (const float*)d_in[21]; const float* bout = (const float*)d_in[22];
    const float* wsig = (const float*)d_in[23]; const float* bsig = (const float*)d_in[24];

    // ---- workspace carve-up (bump allocator, 256B aligned) ----
    char* wp = (char*)d_ws;
    auto alloc = [&](size_t bytes) -> void* {
        void* r = (void*)wp;
        wp += (bytes + 255) & ~(size_t)255;
        return r;
    };
    // all weights stored TRANSPOSED [N,K] in bf16
    __bf16* wq_t = (__bf16*)alloc((size_t)WL * WD * WHD * 2);
    __bf16* wk_t = (__bf16*)alloc((size_t)WL * WD * WHD * 2);
    __bf16* wv_t = (__bf16*)alloc((size_t)WL * WD * WHD * 2);
    __bf16* wo_t = (__bf16*)alloc((size_t)WL * WHD * WD * 2);
    __bf16* w1_t = (__bf16*)alloc((size_t)WL * WD * WDFF * 2);
    __bf16* w2_t = (__bf16*)alloc((size_t)WL * WDFF * WD * 2);
    float*  x_f  = (float*) alloc((size_t)WM * WD * 4);
    __bf16* x_b  = (__bf16*)alloc((size_t)WM * WD * 2);
    __bf16* q_b  = (__bf16*)alloc((size_t)WM * WHD * 2);   // [B,S,H*DH]
    __bf16* k_b  = (__bf16*)alloc((size_t)WM * WHD * 2);   // [B,S,H*DH]
    __bf16* v_t  = (__bf16*)alloc((size_t)WM * WHD * 2);   // TRANSPOSED [H*DH, B*S]
    __bf16* sc_b = (__bf16*)alloc((size_t)WB * WH * WS_ * WS_ * 2);
    __bf16* o_b  = (__bf16*)alloc((size_t)WM * WHD * 2);   // [B,S,H*DH]
    float*  y_f  = (float*) alloc((size_t)WM * WD * 4);
    __bf16* ff_b = (__bf16*)alloc((size_t)WM * WDFF * 2);

    // ---- transpose+convert all weights to bf16 [N,K] (once per launch) ----
    dim3 tb(32, 8);
    transpose_f32_to_bf16<<<dim3(WHD / 32, WD / 32, WL), tb, 0, stream>>>(wq, wq_t, WD, WHD);
    transpose_f32_to_bf16<<<dim3(WHD / 32, WD / 32, WL), tb, 0, stream>>>(wk, wk_t, WD, WHD);
    transpose_f32_to_bf16<<<dim3(WHD / 32, WD / 32, WL), tb, 0, stream>>>(wv, wv_t, WD, WHD);
    transpose_f32_to_bf16<<<dim3(WD / 32, WHD / 32, WL), tb, 0, stream>>>(wo, wo_t, WHD, WD);
    transpose_f32_to_bf16<<<dim3(WDFF / 32, WD / 32, WL), tb, 0, stream>>>(w1, w1_t, WD, WDFF);
    transpose_f32_to_bf16<<<dim3(WD / 32, WDFF / 32, WL), tb, 0, stream>>>(w2, w2_t, WDFF, WD);

    // ---- embedding ----
    embed_kernel<<<(WM * WHALF + 255) / 256, 256, 0, stream>>>(
        inputs, itype, efreq, ew, ephase, x_f, x_b);

    const float iscale = 1.f / sqrtf((float)WDH);

    for (int l = 0; l < WL; ++l) {
        const __bf16* Wq = wq_t + (size_t)l * WD * WHD;
        const __bf16* Wk = wk_t + (size_t)l * WD * WHD;
        const __bf16* Wv = wv_t + (size_t)l * WD * WHD;
        const __bf16* Wo = wo_t + (size_t)l * WHD * WD;
        const __bf16* W1 = w1_t + (size_t)l * WD * WDFF;
        const __bf16* W2 = w2_t + (size_t)l * WDFF * WD;

        // Q,K projections -> [B*S, H*DH]; V projection stored transposed [H*DH, B*S]
        launch_gemm(stream, x_b, WD, 0, 0, 1, Wq, WD, 0, 0, 1,
                    bq + (size_t)l * WHD, nullptr, q_b, WHD, 0, 0, 1, 0,
                    WM, WHD, WD, 1, 1.f, 0);
        launch_gemm(stream, x_b, WD, 0, 0, 1, Wk, WD, 0, 0, 1,
                    bk + (size_t)l * WHD, nullptr, k_b, WHD, 0, 0, 1, 0,
                    WM, WHD, WD, 1, 1.f, 0);
        launch_gemm(stream, x_b, WD, 0, 0, 1, Wv, WD, 0, 0, 1,
                    bv + (size_t)l * WHD, nullptr, v_t, WM, 0, 0, 1, /*cT=*/1,
                    WM, WHD, WD, 1, 1.f, 0);

        // scores[z=b*H+h] = (Q_slice x K_slice^T) * 1/sqrt(DH), batched over 32
        launch_gemm(stream,
                    q_b, WHD, (long)WS_ * WHD, WDH, WH,
                    k_b, WHD, (long)WS_ * WHD, WDH, WH,
                    nullptr, nullptr, sc_b, WS_, (long)WS_ * WS_, 0, 1, 0,
                    WS_, WS_, WDH, WB * WH, iscale, 0);

        softmax_rows<<<WB * WH * WS_, 128, 0, stream>>>(sc_b, WS_);

        // O_slice = probs x V_slice ; V comes from v_t [h*DH+d, b*S+s] => NT operand
        launch_gemm(stream,
                    sc_b, WS_, (long)WS_ * WS_, 0, 1,
                    v_t, WM, (long)WS_, (long)WDH * WM, WH,
                    nullptr, nullptr, o_b, WHD, (long)WS_ * WHD, WDH, WH, 0,
                    WS_, WDH, WS_, WB * WH, 1.f, 0);

        // out projection: [4096,4096] x [4096,512] -> f32 y
        launch_gemm(stream, o_b, WHD, 0, 0, 1, Wo, WHD, 0, 0, 1,
                    bo + (size_t)l * WD, y_f, nullptr, WD, 0, 0, 1, 0,
                    WM, WD, WHD, 1, 1.f, 0);

        add_ln_kernel<<<WM, 128, 0, stream>>>(x_f, x_b, y_f,
                                              ln1g + (size_t)l * WD,
                                              ln1b + (size_t)l * WD, ALPHA_RES);

        // FFN: relu(x W1 + b1) W2 + b2
        launch_gemm(stream, x_b, WD, 0, 0, 1, W1, WD, 0, 0, 1,
                    b1 + (size_t)l * WDFF, nullptr, ff_b, WDFF, 0, 0, 1, 0,
                    WM, WDFF, WD, 1, 1.f, /*relu=*/1);
        launch_gemm(stream, ff_b, WDFF, 0, 0, 1, W2, WDFF, 0, 0, 1,
                    b2 + (size_t)l * WD, y_f, nullptr, WD, 0, 0, 1, 0,
                    WM, WD, WDFF, 1, 1.f, 0);

        add_ln_kernel<<<WM, 128, 0, stream>>>(x_f, x_b, y_f,
                                              ln2g + (size_t)l * WD,
                                              ln2b + (size_t)l * WD, ALPHA_RES);
    }

    head_kernel<<<WM, 64, 0, stream>>>(x_f, wout, bout, wsig, bsig, (float*)d_out);
}
